// FP8QDQSDPA_62070867362249
// MI455X (gfx1250) — 8-bit matrix operand order repaired and checked by reading, compiled, not run on hardware
//
#include <hip/hip_runtime.h>
#include <stdint.h>


#define B_   4
#define S_   1024
#define H_   16
#define D_   64
#define BH_  64
#define SCALE1 0.28125f
#define SCALE2 2.25f
#define INV_QS 0.6666666666666667f
#define F8MAX  448.0f

typedef int    v8i  __attribute__((ext_vector_type(8)));
typedef int    v16i __attribute__((ext_vector_type(16)));
typedef float  v8f  __attribute__((ext_vector_type(8)));

#if __has_builtin(__builtin_amdgcn_cvt_pk_fp8_f32)
#define PK_LO(a,b)     ((unsigned)__builtin_amdgcn_cvt_pk_fp8_f32((a),(b),0,false))
#define PK_HI(a,b,old) ((unsigned)__builtin_amdgcn_cvt_pk_fp8_f32((a),(b),(int)(old),true))
__device__ __forceinline__ unsigned pack4_fp8(float a,float b,float c,float d){
    unsigned r = PK_LO(a,b);
    r = PK_HI(c,d,r);
    return r;
}
__device__ __forceinline__ unsigned fp8_byte(float a){ return PK_LO(a,a) & 0xFFu; }
#else
__device__ __forceinline__ unsigned fp8_byte(float x){
    union { float f; unsigned u; } v; v.f = x;
    unsigned sign = (v.u >> 31) << 7;
    float ax = fabsf(x);
    if (!(ax > 0.0f)) return sign;
    if (ax >= 448.0f) return sign | 0x7Eu;
    if (ax < 0.015625f) {
        int q = (int)rintf(ax * 512.0f);
        if (q > 8) q = 8;
        return sign | (unsigned)q;
    }
    unsigned u = v.u & 0x7FFFFFFFu;
    unsigned lsb = (u >> 20) & 1u;
    u += 0x0007FFFFu + lsb;
    int ec = (int)(u >> 23) - 127 + 7;
    unsigned m = (u >> 20) & 7u;
    if (ec <= 0)  return sign | 0x08u;
    if (ec > 15 || (ec == 15 && m > 6u)) return sign | 0x7Eu;
    return sign | ((unsigned)ec << 3) | m;
}
__device__ __forceinline__ unsigned pack4_fp8(float a,float b,float c,float d){
    return fp8_byte(a) | (fp8_byte(b) << 8) | (fp8_byte(c) << 16) | (fp8_byte(d) << 24);
}
#endif

__device__ __forceinline__ float clampq(float x){
    return fminf(fmaxf(x * INV_QS, -F8MAX), F8MAX);
}

#if __has_builtin(__builtin_amdgcn_tensor_load_to_lds) && __has_builtin(__builtin_amdgcn_s_wait_tensorcnt)
#define HAVE_TDM 1
typedef unsigned int u32x4v __attribute__((ext_vector_type(4)));
typedef int          i32x4v __attribute__((ext_vector_type(4)));
typedef int          i32x8v __attribute__((ext_vector_type(8)));

#if __has_include(<hip/amd_detail/amd_gfx1250_TDM.h>)
#define TDM_ISSUE(g0,g1) do { i32x4v z4_ = {0,0,0,0}; i32x8v z8_ = {0,0,0,0,0,0,0,0}; \
    __builtin_amdgcn_tensor_load_to_lds((g0),(g1),z4_,z4_,z8_,0); } while(0)
#else
#define TDM_ISSUE(g0,g1) do { i32x4v z4_ = {0,0,0,0}; \
    __builtin_amdgcn_tensor_load_to_lds((g0),(g1),z4_,z4_,0); } while(0)
#endif

__device__ __forceinline__ void tdm_load(const void* g, unsigned lds_off,
                                         unsigned tile0, unsigned tile1,
                                         unsigned t0, unsigned t1, unsigned stride0)
{
    unsigned long long ga = (unsigned long long)(uintptr_t)g;
    u32x4v g0;
    g0[0] = 1u;
    g0[1] = lds_off;
    g0[2] = (unsigned)ga;
    g0[3] = (unsigned)((ga >> 32) & 0x01FFFFFFull) | (2u << 30);
    i32x8v g1;
    g1[0] = (int)(3u << 16);
    g1[1] = (int)((t0 & 0xFFFFu) << 16);
    g1[2] = (int)(((t0 >> 16) & 0xFFFFu) | ((t1 & 0xFFFFu) << 16));
    g1[3] = (int)(((t1 >> 16) & 0xFFFFu) | ((tile0 & 0xFFFFu) << 16));
    g1[4] = (int)(tile1 & 0xFFFFu);
    g1[5] = (int)stride0;
    g1[6] = 0;
    g1[7] = 0;
    TDM_ISSUE(g0, g1);
}
#endif

__global__ __launch_bounds__(256)
void quant_rowmajor_kernel(const float* __restrict__ in, uint8_t* __restrict__ out)
{
    unsigned tid = blockIdx.x * 256u + threadIdx.x;
    unsigned o = tid * 4u;
    unsigned d = o & 63u;
    unsigned s = (o >> 6) & 1023u;
    unsigned h = (o >> 16) & 15u;
    unsigned b = o >> 20;
    const float4 f = *(const float4*)(in + (size_t)(b * S_ + s) * (H_ * D_) + h * D_ + d);
    *(unsigned*)(out + o) =
        pack4_fp8(clampq(f.x), clampq(f.y), clampq(f.z), clampq(f.w));
}

__global__ __launch_bounds__(256)
void quant_transpose_kernel(const float* __restrict__ in, uint8_t* __restrict__ out)
{
    unsigned tid = blockIdx.x * 256u + threadIdx.x;
    unsigned o = tid * 4u;
    unsigned s0 = o & 1023u;
    unsigned d  = (o >> 10) & 63u;
    unsigned h  = (o >> 16) & 15u;
    unsigned b  = o >> 20;
    const float* p = in + (size_t)(b * S_ + s0) * (H_ * D_) + h * D_ + d;
    float x0 = p[0];
    float x1 = p[1 * H_ * D_];
    float x2 = p[2 * H_ * D_];
    float x3 = p[3 * H_ * D_];
    *(unsigned*)(out + o) =
        pack4_fp8(clampq(x0), clampq(x1), clampq(x2), clampq(x3));
}

#define WMMA_F8_64(A,Bm,C)  __builtin_amdgcn_wmma_f32_16x16x64_fp8_fp8((A),(Bm),(short)0,(C),false,false)
#define WMMA_F8_128(A,Bm,C) __builtin_amdgcn_wmma_f32_16x16x128_fp8_fp8((A),(Bm),(short)0,(C),false,false)

__device__ __forceinline__ v8i load_b_tile64(const uint8_t* base)
{
    int4 lo = *(const int4*)(base);
    int4 hi = *(const int4*)(base + 32);
    v8i bm;
    bm[0]=lo.x; bm[1]=lo.y; bm[2]=lo.z; bm[3]=lo.w;
    bm[4]=hi.x; bm[5]=hi.y; bm[6]=hi.z; bm[7]=hi.w;
    return bm;
}

__device__ __forceinline__ v16i load_b_tile128(const uint8_t* base)
{
    v16i r;
    #pragma unroll
    for (int g = 0; g < 4; ++g) {
        int4 q = *(const int4*)(base + g * 32);
        r[4*g+0]=q.x; r[4*g+1]=q.y; r[4*g+2]=q.z; r[4*g+3]=q.w;
    }
    return r;
}

__device__ __forceinline__ void stage_chunk(uint8_t* kbuf, uint8_t* vbuf,
                                            const uint8_t* Kg, const uint8_t* Vg,
                                            int chunk, bool with_v)
{
#if defined(HAVE_TDM)
    if (threadIdx.x < 32u) {
        tdm_load(Kg + (size_t)chunk * 8192, (unsigned)(uintptr_t)kbuf,
                 1024u, 0u, 1024u, 1u, 1024u);
        if (with_v) {
            tdm_load(Vg + (size_t)chunk * 128, (unsigned)(uintptr_t)vbuf,
                     16u, 64u, 128u, 64u, 128u);
        }
        __builtin_amdgcn_s_wait_tensorcnt(0);
    }
#else
    const int t = threadIdx.x;
    #pragma unroll
    for (int rep = 0; rep < 2; ++rep) {
        int idx = t + rep * 256;
        *(int4*)(kbuf + idx * 16) = *(const int4*)(Kg + (size_t)chunk * 8192 + idx * 16);
        if (with_v) {
            int dd = idx >> 3, qq = idx & 7;
            *(int4*)(vbuf + dd * 128 + qq * 16) =
                *(const int4*)(Vg + (size_t)dd * 1024 + (size_t)chunk * 128 + qq * 16);
        }
    }
#endif
}

__global__ __launch_bounds__(256)
void attn_kernel(const uint8_t* __restrict__ qf8, const uint8_t* __restrict__ kf8,
                 const uint8_t* __restrict__ vt8, const float* __restrict__ mask,
                 float* __restrict__ out)
{
    __shared__ __align__(16) uint8_t kbuf[8192];
    __shared__ __align__(16) uint8_t vbuf[8192];
    __shared__ __align__(16) uint8_t pbuf[8][16 * 128];

    const int bh   = blockIdx.x;
    const int b    = bh >> 4;
    const int h    = bh & 15;
    const int wave = threadIdx.x >> 5;
    const int lane = threadIdx.x & 31;
    const int n    = lane & 15;
    const int hlf  = lane >> 4;
    const int qbase = blockIdx.y * 128 + wave * 16;

    const uint8_t* Qg = qf8 + (size_t)bh * (S_ * D_);
    const uint8_t* Kg = kf8 + (size_t)bh * (S_ * D_);
    const uint8_t* Vg = vt8 + (size_t)bh * (S_ * D_);

    v8i a;
    {
        const int row = qbase + n;
        #pragma unroll
        for (int v = 0; v < 8; ++v) {
            int k0 = ((v >> 2) << 5) + ((v & 3) << 2) + (hlf << 4);
            a[v] = *(const int*)(Qg + row * 64 + k0);
        }
    }

    float m[8], l[8];
    #pragma unroll
    for (int v = 0; v < 8; ++v) { m[v] = -1e30f; l[v] = 0.0f; }

    for (int chunk = 0; chunk < 8; ++chunk) {
        __syncthreads();
        stage_chunk(kbuf, vbuf, Kg, Vg, chunk, false);
        __syncthreads();
        #pragma unroll
        for (int jj = 0; jj < 8; ++jj) {
            v8i bm = load_b_tile64(kbuf + ((jj * 16 + n) * 64) + (hlf << 4));
            v8f c = 0;
            c = WMMA_F8_64(a, bm, c);
            #pragma unroll
            for (int v = 0; v < 8; ++v) {
                float s  = c[v] * SCALE1;
                float mn = fmaxf(m[v], s);
                l[v] = l[v] * __expf(m[v] - mn) + __expf(s - mn);
                m[v] = mn;
            }
        }
    }

    #pragma unroll
    for (int v = 0; v < 8; ++v) {
        #pragma unroll
        for (int off = 1; off < 16; off <<= 1) {
            float m2 = __shfl_xor(m[v], off, 32);
            float l2 = __shfl_xor(l[v], off, 32);
            float mn = fmaxf(m[v], m2);
            l[v] = l[v] * __expf(m[v] - mn) + l2 * __expf(m2 - mn);
            m[v] = mn;
        }
        l[v] = __builtin_amdgcn_rcpf(l[v]);
    }

    v8f acc[4];
    #pragma unroll
    for (int t = 0; t < 4; ++t) acc[t] = 0;

    uint8_t* pb = pbuf[wave];

    for (int chunk = 0; chunk < 8; ++chunk) {
        __syncthreads();
        stage_chunk(kbuf, vbuf, Kg, Vg, chunk, true);
        __syncthreads();

        if (chunk < 7) {
            __builtin_prefetch(&mask[(size_t)(b * S_ + qbase + lane % 16 + (lane >> 4) * 8) * S_
                                     + (chunk + 1) * 128], 0, 1);
        }

        #pragma unroll
        for (int jj = 0; jj < 8; ++jj) {
            v8i bm = load_b_tile64(kbuf + ((jj * 16 + n) * 64) + (hlf << 4));
            v8f c = 0;
            c = WMMA_F8_64(a, bm, c);
            const int col = chunk * 128 + jj * 16 + n;
            #pragma unroll
            for (int v = 0; v < 8; ++v) {
                const int r = v + hlf * 8;
                float s = c[v] * SCALE1;
                float p = __expf(s - m[v]) * l[v];
                float w = p + mask[(size_t)(b * S_ + (qbase + r)) * S_ + col];
                pb[r * 128 + jj * 16 + n] = (uint8_t)fp8_byte(clampq(w));
            }
        }

        asm volatile("s_wait_dscnt 0" ::: "memory");

        v16i pa;
        #pragma unroll
        for (int v = 0; v < 16; ++v) {
            int vv = v & 7;
            int k0 = ((v >> 2) << 5) + ((v & 3) << 2) + (hlf << 4);
            pa[v] = *(const int*)(pb + n * 128 + k0);
        }

        #pragma unroll
        for (int t = 0; t < 4; ++t) {
            v16i bv = load_b_tile128(vbuf + ((t * 16 + n) * 128) + (hlf << 4));
            acc[t] = WMMA_F8_128(pa, bv, acc[t]);
        }
    }

    #pragma unroll
    for (int t = 0; t < 4; ++t) {
        #pragma unroll
        for (int v = 0; v < 8; ++v) {
            const int r    = v + hlf * 8;
            const int srow = qbase + r;
            const int dcol = t * 16 + n;
            out[(size_t)(b * S_ + srow) * (H_ * D_) + h * D_ + dcol] = acc[t][v] * SCALE2;
        }
    }
}

extern "C" void kernel_launch(void* const* d_in, const int* in_sizes, int n_in,
                              void* d_out, int out_size, void* d_ws, size_t ws_size,
                              hipStream_t stream)
{
    (void)in_sizes; (void)n_in; (void)out_size; (void)ws_size;
    const float* q_in  = (const float*)d_in[0];
    const float* k_in  = (const float*)d_in[1];
    const float* v_in  = (const float*)d_in[2];
    const float* mask  = (const float*)d_in[3];
    float* out = (float*)d_out;

    uint8_t* ws  = (uint8_t*)d_ws;
    uint8_t* Qf8 = ws;
    uint8_t* Kf8 = ws + (4u << 20);
    uint8_t* Vt8 = ws + (8u << 20);

    const int NB = (B_ * H_ * S_ * D_ / 4) / 256;
    quant_rowmajor_kernel<<<NB, 256, 0, stream>>>(v_in, Qf8);
    quant_rowmajor_kernel<<<NB, 256, 0, stream>>>(q_in, Kf8);
    quant_transpose_kernel<<<NB, 256, 0, stream>>>(k_in, Vt8);

    attn_kernel<<<dim3(BH_, S_ / 128), 256, 0, stream>>>(Qf8, Kf8, Vt8, mask, out);
}
